// Decoder_17652315586721
// MI455X (gfx1250) — matrix operand order repaired and checked by reading, not compiled, not run on hardware
//
#include <hip/hip_runtime.h>
#include <hip/hip_bf16.h>

// Problem constants
#define LSEQ 4096
#define HDIM 512
#define MMEM 1024
#define G4   2048   // 4*HDIM

typedef __attribute__((ext_vector_type(16))) __bf16 v16bf;
typedef __attribute__((ext_vector_type(8)))  float  v8f;

// Explicit global (AS1) pointers so VMEM lowers to global_load/global_store
// (LOADcnt/STOREcnt only) instead of flat_load (LOADcnt+DScnt).
typedef const __attribute__((address_space(1))) float cgfloat;
typedef __attribute__((address_space(1))) float gfloat;

// gfx1250 async global->LDS path (present on this toolchain; param types per
// hipcc diagnostic: (AS1 int*, AS3 int*, imm offset, imm cpol)).
#if defined(__has_builtin)
#if __has_builtin(__builtin_amdgcn_global_load_async_to_lds_b32) && \
    __has_builtin(__builtin_amdgcn_s_wait_asynccnt)
#define HAVE_ASYNC_LDS 1
#endif
#endif
#ifndef HAVE_ASYNC_LDS
#define HAVE_ASYNC_LDS 0
#endif

#if HAVE_ASYNC_LDS
typedef __attribute__((address_space(1))) int as1_int;
typedef __attribute__((address_space(3))) int as3_int;
#endif

// Native bf16 convert (RNE) -> v_cvt_pk_bf16_f32 lowering (confirmed round 3).
__device__ inline __bf16 f32_to_bf16(float f) { return (__bf16)f; }

// ---------------------------------------------------------------------------
// 1) Embedding gather: x[l,:] = embed[sent[l],:]
// ---------------------------------------------------------------------------
__global__ void embed_kernel(const int* __restrict__ sent,
                             const float* __restrict__ embed,
                             float* __restrict__ x) {
  const int l = blockIdx.x;
  const size_t row = (size_t)sent[l] * HDIM;
  for (int h = threadIdx.x; h < HDIM; h += blockDim.x)
    x[(size_t)l * HDIM + h] = embed[row + h];
}

// ---------------------------------------------------------------------------
// 2) init: fused bias (b_ih+b_hh), h0/c0 -> state buffers (h double-buffered)
// ---------------------------------------------------------------------------
__global__ void init_state(const float* __restrict__ h0, const float* __restrict__ c0,
                           const float* __restrict__ b_ih, const float* __restrict__ b_hh,
                           float* __restrict__ hbuf, float* __restrict__ cbuf,
                           float* __restrict__ bias) {
  const int j = blockIdx.x * blockDim.x + threadIdx.x;   // 2048 threads
  if (j < G4)  bias[j] = b_ih[j] + b_hh[j];
  if (j < HDIM) { hbuf[j] = h0[j]; cbuf[j] = c0[j]; }
}

// ---------------------------------------------------------------------------
// 3) bf16 WMMA GEMM: D = Am[Mr,K] * B (+ Cadd) (+ bias)
//    BTRANS=true : B[k][n] = Bm[n*ldb+k]  (weight rows, [N,K] row-major)
//    BTRANS=false: B[k][n] = Bm[k*ldb+n]  ([K,N] row-major)
//    Each wave owns a 16 x (16*NT) output strip; A converted once per 32-deep
//    k-chunk, reused across NT independent WMMA chains. Loop-carried AS1
//    pointers (constant bump) -> pure global_load path, no spills.
// ---------------------------------------------------------------------------
template <int NT, bool BTRANS, bool HAS_BIAS, bool HAS_CADD>
__global__ void wmma_gemm(const float* __restrict__ Am, const float* __restrict__ Bm,
                          const float* __restrict__ Cadd, const float* __restrict__ bias,
                          float* __restrict__ D,
                          int N, int K, int lda, int ldb, int ldd) {
  const int lane = threadIdx.x & 31;
  const int wave = threadIdx.x >> 5;
  const int tgN  = N / (16 * NT);
  const int tile = blockIdx.x * 8 + wave;
  const int i0   = (tile / tgN) << 4;
  const int j0   = (tile % tgN) * (16 * NT);

  const int half = lane >> 4;          // K-half selector
  const int mrow = lane & 15;          // A row within tile
  const int nn   = lane & 15;          // B/D column within tile
  const int kb   = half << 4;          // B-operand K base

  // loop-carried per-lane global pointers
  cgfloat* ap = (cgfloat*)(Am + (size_t)(i0 + mrow) * lda);      // +32/iter
  const float* apg = Am + (size_t)(i0 + mrow) * lda;             // prefetch shadow
  cgfloat* bp[NT];
#pragma unroll
  for (int s = 0; s < NT; ++s) {
    const int jn = j0 + s * 16 + nn;
    bp[s] = BTRANS ? (cgfloat*)(Bm + (size_t)jn * ldb + kb)      // +32/iter
                   : (cgfloat*)(Bm + (size_t)kb * ldb + jn);     // +32*ldb/iter
  }
  const size_t bstep = BTRANS ? 32 : (size_t)32 * ldb;

  v8f zero = {};
  v8f acc[NT];
#pragma unroll
  for (int s = 0; s < NT; ++s) acc[s] = zero;

  for (int k = 0; k < K; k += 32) {
    __builtin_prefetch(apg + 32, 0, 1);   // global_prefetch_b8
    apg += 32;
    v16bf a;
#pragma unroll
    for (int e = 0; e < 16; ++e) {
      const int ka = (half << 4) + e;    // contiguous K map (k = 16*half + e), matches B
      a[e] = f32_to_bf16(ap[ka]);
    }
    ap += 32;
#pragma unroll
    for (int s = 0; s < NT; ++s) {
      v16bf b;
      if (BTRANS) {
#pragma unroll
        for (int e = 0; e < 16; ++e) b[e] = f32_to_bf16(bp[s][e]);
      } else {
#pragma unroll
        for (int e = 0; e < 16; ++e) b[e] = f32_to_bf16(bp[s][(size_t)e * ldb]);
      }
      bp[s] += bstep;
      acc[s] = __builtin_amdgcn_wmma_f32_16x16x32_bf16(false, a, false, b,
                                                       (short)0, acc[s], false, false);
    }
  }

#pragma unroll
  for (int s = 0; s < NT; ++s) {
#pragma unroll
    for (int r = 0; r < 8; ++r) {
      const int m = i0 + r + (half << 3);     // ISA C/D map
      const int n = j0 + s * 16 + nn;
      float v = acc[s][r];
      if (HAS_BIAS) v += ((cgfloat*)bias)[n];
      if (HAS_CADD) v += ((cgfloat*)Cadd)[(size_t)m * ldd + n];
      ((gfloat*)D)[(size_t)m * ldd + n] = v;
    }
  }
}

// ---------------------------------------------------------------------------
// 4) Column softmax over the sequence dim (axis=1 of [1,L,M])
// ---------------------------------------------------------------------------
__global__ void softmax_col(float* __restrict__ sc, int L, int M) {
  __shared__ float red[256];
  const int m = blockIdx.x, tid = threadIdx.x;

  float mv = -3.402823466e38f;
  for (int l = tid; l < L; l += 256) mv = fmaxf(mv, sc[(size_t)l * M + m]);
  red[tid] = mv; __syncthreads();
  for (int s = 128; s > 0; s >>= 1) { if (tid < s) red[tid] = fmaxf(red[tid], red[tid + s]); __syncthreads(); }
  const float cmax = red[0]; __syncthreads();

  float sv = 0.f;
  for (int l = tid; l < L; l += 256) sv += __expf(sc[(size_t)l * M + m] - cmax);
  red[tid] = sv; __syncthreads();
  for (int s = 128; s > 0; s >>= 1) { if (tid < s) red[tid] += red[tid + s]; __syncthreads(); }
  const float inv = 1.f / red[0];

  for (int l = tid; l < L; l += 256)
    sc[(size_t)l * M + m] = __expf(sc[(size_t)l * M + m] - cmax) * inv;
}

// ---------------------------------------------------------------------------
// 5) One LSTM step (fp32, launch-per-step = grid-wide sync).
//    8 blocks x 256 threads; block owns 64 hidden units -> all 4 gates local.
//    h staged into LDS via gfx1250 async global->LDS path.
// ---------------------------------------------------------------------------
__global__ void lstm_step(const float* __restrict__ G, const float* __restrict__ W_hh,
                          float* __restrict__ hbuf, float* __restrict__ cbuf,
                          float* __restrict__ outputs, int t) {
  __shared__ float hs[HDIM];
  __shared__ float gate_s[256];
  const int tid  = threadIdx.x;
  const int base = blockIdx.x * 64;                 // hidden chunk [base, base+64)
  const float* h_in  = hbuf + (t & 1) * HDIM;
  float*       h_out = hbuf + ((t + 1) & 1) * HDIM;

#if HAVE_ASYNC_LDS
  for (int i = tid; i < HDIM; i += 256)
    __builtin_amdgcn_global_load_async_to_lds_b32(
        (as1_int*)(h_in + i), (as3_int*)(&hs[i]), 0, 0);
  __builtin_amdgcn_s_wait_asynccnt(0);
  __syncthreads();
#else
  for (int i = tid; i < HDIM; i += 256) hs[i] = h_in[i];
  __syncthreads();
#endif

  const int lane = tid & 31, wave = tid >> 5;
  for (int r = 0; r < 32; ++r) {
    const int o = wave * 32 + r;                    // local output 0..255
    const int g = o >> 6, q = o & 63;
    const int j = g * HDIM + base + q;              // W_hh row / gate index
    cgfloat* wrow = (cgfloat*)(W_hh + (size_t)j * HDIM);
    float p = 0.f;
#pragma unroll
    for (int i = 0; i < HDIM / 32; ++i)
      p += hs[lane + 32 * i] * wrow[lane + 32 * i]; // coalesced across lanes
#pragma unroll
    for (int off = 16; off > 0; off >>= 1)
      p += __shfl_xor(p, off, 32);
    if (lane == 0) gate_s[o] = p + G[(size_t)t * G4 + j];  // pre-gates incl. biases
  }
  __syncthreads();

  if (tid < 64) {
    const int hh = base + tid;
    const float ig = 1.f / (1.f + __expf(-gate_s[0 * 64 + tid]));
    const float fg = 1.f / (1.f + __expf(-gate_s[1 * 64 + tid]));
    const float gg = tanhf(gate_s[2 * 64 + tid]);
    const float og = 1.f / (1.f + __expf(-gate_s[3 * 64 + tid]));
    const float c = fg * cbuf[hh] + ig * gg;
    const float h = og * tanhf(c);
    cbuf[hh] = c;
    h_out[hh] = h;
    outputs[(size_t)t * HDIM + hh] = h;
  }
}

// ---------------------------------------------------------------------------
// 6) finalize: hT, cT, A_new = [A ; hT]
// ---------------------------------------------------------------------------
__global__ void finalize(const float* __restrict__ h_fin, const float* __restrict__ c_fin,
                         const float* __restrict__ A, float* __restrict__ out) {
  const size_t ANEW_OFF = (size_t)1024 + (size_t)LSEQ * HDIM;
  const int total = 1024 + (MMEM + 1) * HDIM;
  for (int i = blockIdx.x * blockDim.x + threadIdx.x; i < total;
       i += gridDim.x * blockDim.x) {
    if (i < HDIM)            out[i] = h_fin[i];
    else if (i < 2 * HDIM)   out[i] = c_fin[i - HDIM];
    else {
      const int k = i - 1024;
      const int r = k / HDIM, hh = k % HDIM;
      out[ANEW_OFF + k] = (r < MMEM) ? A[(size_t)r * HDIM + hh] : h_fin[hh];
    }
  }
}

// ---------------------------------------------------------------------------
extern "C" void kernel_launch(void* const* d_in, const int* in_sizes, int n_in,
                              void* d_out, int out_size, void* d_ws, size_t ws_size,
                              hipStream_t stream) {
  const int*   sent  = (const int*)  d_in[0];
  const float* h0    = (const float*)d_in[1];
  const float* c0    = (const float*)d_in[2];
  const float* A     = (const float*)d_in[3];
  const float* embed = (const float*)d_in[4];
  const float* W_ih  = (const float*)d_in[5];
  const float* W_hh  = (const float*)d_in[6];
  const float* b_ih  = (const float*)d_in[7];
  const float* b_hh  = (const float*)d_in[8];
  float* out = (float*)d_out;
  float* ws  = (float*)d_ws;

  // workspace layout (floats)
  float* x    = ws;                                   // L*H      = 2,097,152
  float* sc   = ws + (size_t)LSEQ * HDIM;             // L*M      = 4,194,304
  float* G    = sc + (size_t)LSEQ * MMEM;             // L*4H     = 8,388,608
  float* hbuf = G  + (size_t)LSEQ * G4;               // 2*H (double-buffered h)
  float* cbuf = hbuf + 2 * HDIM;                      // H
  float* bias = cbuf + HDIM;                          // 4H

  embed_kernel<<<LSEQ, 256, 0, stream>>>(sent, embed, x);
  init_state<<<8, 256, 0, stream>>>(h0, c0, b_ih, b_hh, hbuf, cbuf, bias);

  // scores = x @ A^T   [4096,1024]
  wmma_gemm<4, true, false, false><<<(LSEQ/16)*(MMEM/64)/8, 256, 0, stream>>>(
      x, A, nullptr, nullptr, sc, MMEM, HDIM, HDIM, HDIM, MMEM);

  // softmax over sequence dim, per memory slot (in place -> att)
  softmax_col<<<MMEM, 256, 0, stream>>>(sc, LSEQ, MMEM);

  // x = x + att @ A    [4096,512] (in place: each element read+written by owner lane)
  wmma_gemm<4, false, false, true><<<(LSEQ/16)*(HDIM/64)/8, 256, 0, stream>>>(
      sc, A, x, nullptr, x, HDIM, MMEM, MMEM, HDIM, HDIM);

  // pre-gates for all timesteps: G = x @ W_ih^T + (b_ih + b_hh)   [4096,2048]
  wmma_gemm<4, true, true, false><<<(LSEQ/16)*(G4/64)/8, 256, 0, stream>>>(
      x, W_ih, nullptr, bias, G, G4, HDIM, HDIM, HDIM, G4);

  // sequential recurrence: one launch per step (device-wide dependency chain)
  float* outputs = out + 1024;
  for (int t = 0; t < LSEQ; ++t)
    lstm_step<<<8, 256, 0, stream>>>(G, W_hh, hbuf, cbuf, outputs, t);

  // final h lives in buffer (4096 & 1) == 0
  finalize<<<512, 256, 0, stream>>>(hbuf, cbuf, A, out);
}
